// VarianceAdaptor_89309549953662
// MI455X (gfx1250) — matrix operand order repaired and checked by reading, not compiled, not run on hardware
//
#include <hip/hip_runtime.h>
#include <math.h>

// ---------------- problem constants ----------------
#define BB   32
#define CCH  384
#define ODIMC 80
#define ODIMP 96     // padded Cin for f1
#define TTX  256     // T_TEXT
#define TFR  1024    // T_FEATS
#define KDWC 7
#define NEGV (-1e30f)

// output section offsets (floats)
#define N_XF   (BB*CCH*TFR)
#define O_DUR  (N_XF)
#define O_LDP  (O_DUR + BB*TTX)
#define O_LCF0 (O_LDP + BB*TTX)
#define O_VUV  (O_LCF0 + BB*TFR)
#define O_LB   (O_VUV + BB*TFR)

typedef __attribute__((ext_vector_type(16))) _Float16 v16h;
typedef __attribute__((ext_vector_type(8)))  _Float16 v8h;
typedef __attribute__((ext_vector_type(8)))  float    v8f;

__device__ __forceinline__ float gelu_exact(float x) {
  return 0.5f * x * (1.0f + erff(x * 0.7071067811865475f));
}
__device__ __forceinline__ float lse2(float a, float b) {
  float m = fmaxf(a, b), n = fminf(a, b);
  return m + log1pf(expf(n - m));
}
__device__ __forceinline__ float cleanv(float v) { return fmaxf(v, NEGV); }

// ============================================================
// conv1d-as-GEMM with WMMA. One wave -> one 16x16 tile, K fully unrolled.
// Wh : f16 weights  [tap][Cout][CINP]        (K contiguous)
// Xt : f16 activations, transposed [B][T][CINP] (K contiguous)
// Yf : optional f32 out [B][Cout][T];  Yt : optional f16 out [B][T][Cout]
// act: 0 none, 1 gelu, 2 relu; optional layer-scale, residual R, mask
// ============================================================
template<int CINP, int KW>
__global__ __launch_bounds__(32)
void conv_wmma2(const _Float16* __restrict__ Wh, const float* __restrict__ bias,
                const _Float16* __restrict__ Xt, float* Yf, _Float16* __restrict__ Yt,
                const float* R, const float* __restrict__ lscale,
                const float* __restrict__ mask,
                int Cout, int T, int pad, int act)
{
  const int lane = threadIdx.x & 31;
  const int half = lane >> 4;
  const int ln   = lane & 15;
  const int t0 = blockIdx.x * 16;
  const int o0 = blockIdx.y * 16;
  const int b  = blockIdx.z;

  const _Float16* Xb = Xt + (size_t)b * T * CINP;
  v8f acc = {};

#pragma unroll
  for (int tap = 0; tap < KW; ++tap) {
    const int ts = t0 + ln + tap - pad;
    const bool tok = (ts >= 0) && (ts < T);
    const _Float16* xrow = Xb + (size_t)(tok ? ts : 0) * CINP;
    const _Float16* wrow = Wh + ((size_t)tap * Cout + (o0 + ln)) * CINP;
#pragma unroll
    for (int k0 = 0; k0 < CINP; k0 += 32) {
      // A fragment: row m=ln, two contiguous 8-half runs
      v8h al = *(const v8h*)(wrow + k0 + 8 * half);
      v8h ah = *(const v8h*)(wrow + k0 + 16 + 8 * half);
      v16h av = __builtin_shufflevector(al, ah, 0,1,2,3,4,5,6,7,8,9,10,11,12,13,14,15);
      // B fragment: column n=ln, two contiguous 8-half runs
      v8h bl = {}, bh = {};
      if (tok) {
        bl = *(const v8h*)(xrow + k0 + 8 * half);
        bh = *(const v8h*)(xrow + k0 + 16 + 8 * half);
      }
      v16h bv = __builtin_shufflevector(bl, bh, 0,1,2,3,4,5,6,7,8,9,10,11,12,13,14,15);
      acc = __builtin_amdgcn_wmma_f32_16x16x32_f16(false, av, false, bv,
                                                   (short)0, acc, false, false);
    }
  }

  const int t = t0 + ln;
  const float mk = mask ? mask[(size_t)b * T + t] : 1.f;
  float vr[8];
#pragma unroll
  for (int e = 0; e < 8; ++e) {
    const int o = o0 + e + 8 * half;
    float v = acc[e] + bias[o];
    if (act == 1)      v = gelu_exact(v);
    else if (act == 2) v = fmaxf(v, 0.f);
    if (lscale) v *= lscale[o];
    if (R)      v += R[((size_t)b * Cout + o) * T + t];
    v *= mk;
    vr[e] = v;
    if (Yf) Yf[((size_t)b * Cout + o) * T + t] = v;
  }
  if (Yt) {
    v8h pk;
#pragma unroll
    for (int e = 0; e < 8; ++e) pk[e] = (_Float16)vr[e];
    *(v8h*)(Yt + ((size_t)b * T + t) * Cout + o0 + 8 * half) = pk;
  }
}

// ============================================================
// FT[b][j][i] = sum_c f[b][c][j] * t[b][c][i]
// Fjc: f16 [B][TFR][CCH] (f transposed), Tic: f16 [B][TTX][CCH] (t transposed)
// ============================================================
__global__ __launch_bounds__(32)
void ft_wmma2(const _Float16* __restrict__ Fjc, const _Float16* __restrict__ Tic,
              float* __restrict__ FT)
{
  const int lane = threadIdx.x & 31;
  const int half = lane >> 4;
  const int ln   = lane & 15;
  const int i0 = blockIdx.x * 16;
  const int j0 = blockIdx.y * 16;
  const int b  = blockIdx.z;

  const _Float16* arow = Fjc + ((size_t)b * TFR + (j0 + ln)) * CCH;
  const _Float16* brow = Tic + ((size_t)b * TTX + (i0 + ln)) * CCH;
  v8f acc = {};
#pragma unroll
  for (int k0 = 0; k0 < CCH; k0 += 32) {
    v8h al = *(const v8h*)(arow + k0 + 8 * half);
    v8h ah = *(const v8h*)(arow + k0 + 16 + 8 * half);
    v16h av = __builtin_shufflevector(al, ah, 0,1,2,3,4,5,6,7,8,9,10,11,12,13,14,15);
    v8h bl = *(const v8h*)(brow + k0 + 8 * half);
    v8h bh = *(const v8h*)(brow + k0 + 16 + 8 * half);
    v16h bv = __builtin_shufflevector(bl, bh, 0,1,2,3,4,5,6,7,8,9,10,11,12,13,14,15);
    acc = __builtin_amdgcn_wmma_f32_16x16x32_f16(false, av, false, bv,
                                                 (short)0, acc, false, false);
  }
#pragma unroll
  for (int e = 0; e < 8; ++e)
    FT[((size_t)b * TFR + (j0 + e + 8 * half)) * TTX + (i0 + ln)] = acc[e];
}

// ============================================================
// tiled LDS transpose + f32->f16 convert: X[B][Cin][T] -> Xt[B][T][Cpad]
// ============================================================
__global__ __launch_bounds__(256)
void transpose_f16_kernel(const float* __restrict__ X, _Float16* __restrict__ Xt,
                          int Cin, int T, int Cpad)
{
  __shared__ float tile[16][17];
  const int t0 = blockIdx.x * 16, c0 = blockIdx.y * 16, b = blockIdx.z;
  const int tx = threadIdx.x & 15, ty = threadIdx.x >> 4;
  const int c = c0 + ty;
  tile[ty][tx] = (c < Cin) ? X[((size_t)b * Cin + c) * T + (t0 + tx)] : 0.f;
  __syncthreads();
  Xt[((size_t)b * T + (t0 + ty)) * Cpad + (c0 + tx)] = (_Float16)tile[tx][ty];
}

// weights f32 [O][I][kw] -> f16 [tap][O][Ipad]
__global__ void wconv_kernel(const float* __restrict__ W, _Float16* __restrict__ Wh,
                             int O, int I, int kw, int Ipad)
{
  const int idx = blockIdx.x * blockDim.x + threadIdx.x;
  if (idx >= kw * O * Ipad) return;
  const int i = idx % Ipad, o = (idx / Ipad) % O, tap = idx / (Ipad * O);
  Wh[idx] = (_Float16)((i < I) ? W[((size_t)o * I + i) * kw + tap] : 0.f);
}

// ============================================================
// depthwise conv(K=7) + bias + channel LayerNorm + affine, per (b,t)
// writes f16 transposed [B][T][CCH] (feeds pw1 WMMA directly)
// ============================================================
__global__ __launch_bounds__(CCH)
void dwln_kernel(const float* __restrict__ X, const float* __restrict__ dww,
                 const float* __restrict__ dwb, const float* __restrict__ g,
                 const float* __restrict__ bt, _Float16* __restrict__ Ht, int T)
{
  const int b = blockIdx.x / T, t = blockIdx.x % T, c = threadIdx.x;
  float s = dwb[c];
#pragma unroll
  for (int k = 0; k < KDWC; ++k) {
    const int ts = t + k - (KDWC - 1) / 2;
    if (ts >= 0 && ts < T) s += dww[c * KDWC + k] * X[((size_t)b * CCH + c) * T + ts];
  }
  __shared__ float red[512];
  red[c] = s; if (c < 128) red[CCH + c] = 0.f;
  __syncthreads();
  for (int off = 256; off > 0; off >>= 1) { if (c < off) red[c] += red[c + off]; __syncthreads(); }
  const float mu = red[0] / (float)CCH;
  __syncthreads();
  const float d = s - mu;
  red[c] = d * d; if (c < 128) red[CCH + c] = 0.f;
  __syncthreads();
  for (int off = 256; off > 0; off >>= 1) { if (c < off) red[c] += red[c + off]; __syncthreads(); }
  const float var = red[0] / (float)CCH;
  Ht[((size_t)b * T + t) * CCH + c] = (_Float16)(d * rsqrtf(var + 1e-5f) * g[c] + bt[c]);
}

// small out-conv (Cout=1 or 2), fused mask
__global__ void outconv_kernel(const float* __restrict__ W, const float* __restrict__ bias,
                               const float* __restrict__ X, float* __restrict__ Y,
                               const float* __restrict__ mask, int Cout, int T)
{
  const int idx = blockIdx.x * blockDim.x + threadIdx.x;
  if (idx >= BB * Cout * T) return;
  const int t = idx % T, o = (idx / T) % Cout, b = idx / (T * Cout);
  float s = bias[o];
  for (int c = 0; c < CCH; ++c) s += W[o * CCH + c] * X[((size_t)b * CCH + c) * T + t];
  Y[idx] = s * mask[(size_t)b * T + t];
}

__global__ void sumsq_kernel(const float* __restrict__ X, float* __restrict__ S, int T)
{
  const int idx = blockIdx.x * blockDim.x + threadIdx.x;
  if (idx >= BB * T) return;
  const int b = idx / T, t = idx % T;
  float s = 0.f;
  for (int c = 0; c < CCH; ++c) { float v = X[((size_t)b * CCH + c) * T + t]; s += v * v; }
  S[idx] = s;
}

// score = -sqrt(max(d2,1e-12)), text-mask, row log-softmax over i (256)
__global__ __launch_bounds__(TTX)
void score_kernel(const float* __restrict__ FT, const float* __restrict__ f2s,
                  const float* __restrict__ t2s, const int* __restrict__ xlen,
                  float* __restrict__ LOGP)
{
  const int b = blockIdx.x / TFR, j = blockIdx.x % TFR, i = threadIdx.x;
  const float ftv = FT[((size_t)b * TFR + j) * TTX + i];
  float d2 = f2s[b * TFR + j] + t2s[b * TTX + i] - 2.f * ftv;
  float sc = -sqrtf(fmaxf(d2, 1e-12f));
  const bool valid = i < xlen[b];
  sc = valid ? sc : -INFINITY;
  __shared__ float red[TTX];
  red[i] = sc; __syncthreads();
  for (int off = 128; off > 0; off >>= 1) { if (i < off) red[i] = fmaxf(red[i], red[i + off]); __syncthreads(); }
  const float m = red[0];
  __syncthreads();
  red[i] = valid ? expf(sc - m) : 0.f; __syncthreads();
  for (int off = 128; off > 0; off >>= 1) { if (i < off) red[i] += red[i + off]; __syncthreads(); }
  LOGP[((size_t)b * TFR + j) * TTX + i] = sc - (m + logf(red[0]));
}

// Viterbi forward scan over frames, Q stored for backtrack
__global__ __launch_bounds__(TTX)
void vit_fwd_kernel(const float* __restrict__ LOGP, float* __restrict__ Q)
{
  const int b = blockIdx.x, i = threadIdx.x;
  __shared__ float q[TTX];
  float q0 = (i == 0) ? cleanv(LOGP[((size_t)b * TFR) * TTX]) : NEGV;
  q[i] = q0; Q[((size_t)b * TFR) * TTX + i] = q0;
  __syncthreads();
  for (int f = 1; f < TFR; ++f) {
    const float qi = q[i];
    const float qm = (i > 0) ? q[i - 1] : NEGV;
    __syncthreads();
    const float qn = cleanv(LOGP[((size_t)b * TFR + f) * TTX + i]) + fmaxf(qi, qm);
    q[i] = qn; Q[((size_t)b * TFR + f) * TTX + i] = qn;
    __syncthreads();
  }
}

__global__ void vit_back_kernel(const float* __restrict__ Q, const int* __restrict__ xlen,
                                const int* __restrict__ ylen, int* __restrict__ A)
{
  const int b = threadIdx.x;
  if (b >= BB) return;
  const int tx = xlen[b], ty = ylen[b];
  int an = tx - 1;
  for (int j = TFR - 1; j >= 0; --j) {
    const float* Qj = Q + ((size_t)b * TFR + j) * TTX;
    const int ib = an - 1;
    int cand = (ib < 0) ? an : ((Qj[an] >= Qj[ib]) ? an : ib);
    const int a = (j >= ty - 1) ? (tx - 1) : cand;
    A[b * TFR + j] = a;
    an = a;
  }
}

__global__ void dur_kernel(const int* __restrict__ A, const int* __restrict__ ylen,
                           float* __restrict__ dur)
{
  const int idx = blockIdx.x * blockDim.x + threadIdx.x;
  if (idx >= BB * TTX) return;
  const int b = idx / TTX, tt = idx % TTX, ty = ylen[b];
  float c = 0.f;
  for (int j = 0; j < ty; ++j) if (A[b * TFR + j] == tt) c += 1.f;
  dur[idx] = c;
}

__global__ void binacc_kernel(const int* __restrict__ A, const float* __restrict__ LOGP,
                              const int* __restrict__ ylen, float* __restrict__ binacc)
{
  const int b = threadIdx.x;
  if (b >= BB) return;
  const int ty = ylen[b];
  float s = 0.f;
  for (int j = 0; j < ty; ++j)
    s += LOGP[((size_t)b * TFR + j) * TTX + A[b * TFR + j]];
  binacc[b] = s;
}

// CTC emission table: log-softmax over 257 (blank + 256 text cols, masked by tx)
__global__ __launch_bounds__(TTX)
void lls_kernel(const float* __restrict__ LOGP, const int* __restrict__ xlen,
                float* __restrict__ LLS)
{
  const int b = blockIdx.x / TFR, t = blockIdx.x % TFR, i = threadIdx.x;
  const int tx = xlen[b];
  const int c = i + 1;
  const float v = (c <= tx) ? LOGP[((size_t)b * TFR + t) * TTX + i] : NEGV;
  __shared__ float red[TTX];
  red[i] = v; __syncthreads();
  for (int off = 128; off > 0; off >>= 1) { if (i < off) red[i] = fmaxf(red[i], red[i + off]); __syncthreads(); }
  const float m = fmaxf(red[0], -1.0f);
  __syncthreads();
  red[i] = expf(v - m); __syncthreads();
  for (int off = 128; off > 0; off >>= 1) { if (i < off) red[i] += red[i + off]; __syncthreads(); }
  const float lse = m + logf(red[0] + expf(-1.0f - m));
  LLS[((size_t)b * TFR + t) * 257 + c] = v - lse;
  if (i == 0) LLS[((size_t)b * TFR + t) * 257] = -1.0f - lse;
}

// CTC alpha scan: S=513 states, block per batch
__global__ __launch_bounds__(544)
void ctc_kernel(const float* __restrict__ LLS, const int* __restrict__ xlen,
                const int* __restrict__ ylen, float* __restrict__ ctcl)
{
  const int b = blockIdx.x, s = threadIdx.x;
  const int tx = xlen[b], ty = ylen[b];
  const bool act = s < 513;
  const int lab = (s & 1) ? ((s + 1) >> 1) : 0;
  const float* lbase = LLS + (size_t)b * TFR * 257;
  __shared__ float al[513];
  if (act) al[s] = (s <= 1) ? lbase[lab] : NEGV;
  __syncthreads();
  for (int t = 1; t < TFR; ++t) {
    float a0 = 0.f, a1 = 0.f, a2 = 0.f;
    if (act) { a0 = al[s]; a1 = (s >= 1) ? al[s - 1] : NEGV; a2 = (s >= 2) ? al[s - 2] : NEGV; }
    __syncthreads();
    if (act) {
      float comb = lse2(a0, a1);
      if (s & 1) comb = lse2(comb, a2);
      al[s] = (t < ty) ? (lbase[(size_t)t * 257 + lab] + comb) : a0;
    }
    __syncthreads();
  }
  if (s == 0) {
    const float ll = lse2(al[2 * tx], al[2 * tx - 1]);
    ctcl[b] = (fabsf(ll) <= 3.0e38f) ? (-ll / (float)tx) : 0.f;
  }
}

__global__ void cum_kernel(const float* __restrict__ dur, float* __restrict__ cum)
{
  const int b = threadIdx.x;
  if (b >= BB) return;
  float s = 0.f;
  for (int i = 0; i < TTX; ++i) { s += dur[b * TTX + i]; cum[b * TTX + i] = s; }
}

__global__ void assign_kernel(const float* __restrict__ cum, const float* __restrict__ xmask,
                              const float* __restrict__ ymask, int* __restrict__ idxb)
{
  const int idx = blockIdx.x * blockDim.x + threadIdx.x;
  if (idx >= BB * TFR) return;
  const int b = idx / TFR, f = idx % TFR;
  const float ff = (float)f;
  const float* cb = cum + b * TTX;
  int lo = 0, hi = TTX;
  while (lo < hi) { const int mid = (lo + hi) >> 1; if (cb[mid] > ff) hi = mid; else lo = mid + 1; }
  int r = -1;
  if (lo < TTX) {
    const float pm = xmask[b * TTX + lo] * ymask[b * TFR + f];
    if (pm > 0.5f) r = lo;
  }
  idxb[idx] = r;
}

__global__ void gather_kernel(const float* __restrict__ x, const int* __restrict__ idxb,
                              float* __restrict__ XF, float* __restrict__ XP)
{
  const int idx = blockIdx.x * blockDim.x + threadIdx.x;
  if (idx >= BB * CCH * TFR) return;
  const int f = idx % TFR, c = (idx / TFR) % CCH, b = idx / (TFR * CCH);
  const int tt = idxb[b * TFR + f];
  const float v = (tt >= 0) ? x[((size_t)b * CCH + c) * TTX + tt] : 0.f;
  XF[idx] = v; XP[idx] = v;
}

__global__ void copy_kernel(const float* __restrict__ src, float* __restrict__ dst, int n)
{
  const int idx = blockIdx.x * blockDim.x + threadIdx.x;
  if (idx < n) dst[idx] = src[idx];
}

__global__ void out_xframe_kernel(const float* __restrict__ XF, const float* __restrict__ pew,
                                  const float* __restrict__ peb, const float* __restrict__ lcf0,
                                  const float* __restrict__ ymask, float* __restrict__ out)
{
  const int idx = blockIdx.x * blockDim.x + threadIdx.x;
  if (idx >= BB * CCH * TFR) return;
  const int f = idx % TFR, c = (idx / TFR) % CCH, b = idx / (TFR * CCH);
  const float pitch = (pew[c] * lcf0[b * TFR + f] + peb[c]) * ymask[b * TFR + f];
  out[idx] = XF[idx] + pitch;
}

__global__ void out_small_kernel(const float* __restrict__ dur, const float* __restrict__ dpout,
                                 const float* __restrict__ pred, float* __restrict__ out)
{
  int idx = blockIdx.x * blockDim.x + threadIdx.x;
  if (idx < BB * TTX) { out[O_DUR + idx] = dur[idx]; return; }
  idx -= BB * TTX;
  if (idx < BB * TTX) { out[O_LDP + idx] = dpout[idx]; return; }
  idx -= BB * TTX;
  if (idx < BB * TFR) {
    const int b = idx / TFR, f = idx % TFR;
    out[O_LCF0 + idx] = pred[((size_t)b * 2) * TFR + f];
    return;
  }
  idx -= BB * TFR;
  if (idx < BB * TFR) {
    const int b = idx / TFR, f = idx % TFR;
    const float v = pred[((size_t)b * 2 + 1) * TFR + f];
    out[O_VUV + idx] = 1.f / (1.f + expf(-v));
  }
}

__global__ void loss_kernel(const float* __restrict__ binacc, const int* __restrict__ ylen,
                            const float* __restrict__ ctcl, float* __restrict__ out)
{
  __shared__ float r1[BB], r2[BB];
  const int b = threadIdx.x;
  if (b < BB) { r1[b] = -binacc[b] / (float)ylen[b]; r2[b] = ctcl[b]; }
  __syncthreads();
  if (b == 0) {
    float s1 = 0.f, s2 = 0.f;
    for (int i = 0; i < BB; ++i) { s1 += r1[i]; s2 += r2[i]; }
    out[O_LB] = s1 / (float)BB;
    out[O_LB + 1] = s2 / (float)BB;
  }
}

// ============================================================
extern "C" void kernel_launch(void* const* d_in, const int* in_sizes, int n_in,
                              void* d_out, int out_size, void* d_ws, size_t ws_size,
                              hipStream_t stream) {
  (void)in_sizes; (void)n_in; (void)out_size; (void)ws_size;
  const float* x     = (const float*)d_in[0];
  const float* y     = (const float*)d_in[1];
  const float* lcf0  = (const float*)d_in[2];
  const float* xmask = (const float*)d_in[3];
  const float* ymask = (const float*)d_in[4];
  const int*   xlen  = (const int*)d_in[5];
  const int*   ylen  = (const int*)d_in[6];

  int pi = 7;
  struct LayerP { const float *dww,*dwb,*lng,*lnb,*pw1w,*pw1b,*pw2w,*pw2b,*scale; };
  LayerP dpl[4], ppl[4];
  for (int l = 0; l < 4; ++l) {
    dpl[l] = { (const float*)d_in[pi+0],(const float*)d_in[pi+1],(const float*)d_in[pi+2],
               (const float*)d_in[pi+3],(const float*)d_in[pi+4],(const float*)d_in[pi+5],
               (const float*)d_in[pi+6],(const float*)d_in[pi+7],(const float*)d_in[pi+8] };
    pi += 9;
  }
  const float* dp_ow = (const float*)d_in[pi++]; const float* dp_ob = (const float*)d_in[pi++];
  for (int l = 0; l < 4; ++l) {
    ppl[l] = { (const float*)d_in[pi+0],(const float*)d_in[pi+1],(const float*)d_in[pi+2],
               (const float*)d_in[pi+3],(const float*)d_in[pi+4],(const float*)d_in[pi+5],
               (const float*)d_in[pi+6],(const float*)d_in[pi+7],(const float*)d_in[pi+8] };
    pi += 9;
  }
  const float* pp_ow = (const float*)d_in[pi++]; const float* pp_ob = (const float*)d_in[pi++];
  const float* t1w = (const float*)d_in[pi++]; const float* t1b = (const float*)d_in[pi++];
  const float* t2w = (const float*)d_in[pi++]; const float* t2b = (const float*)d_in[pi++];
  const float* f1w = (const float*)d_in[pi++]; const float* f1b = (const float*)d_in[pi++];
  const float* f2w = (const float*)d_in[pi++]; const float* f2b = (const float*)d_in[pi++];
  const float* f3w = (const float*)d_in[pi++]; const float* f3b = (const float*)d_in[pi++];
  const float* pew = (const float*)d_in[pi++]; const float* peb = (const float*)d_in[pi++];

  // ---- workspace layout (32B-aligned blocks) ----
  float* w = (float*)d_ws;
  size_t off = 0;
  auto alloc  = [&](size_t n) { n = (n + 7) & ~(size_t)7; float* p = w + off; off += n; return p; };
  auto allocH = [&](size_t nh) { return (_Float16*)alloc((nh + 1) / 2); };

  float* XDP  = alloc((size_t)BB*CCH*TTX);
  float* Hb   = alloc((size_t)BB*CCH*TFR);   // f3 f32 out (for f2s)
  float* Tb   = alloc((size_t)BB*CCH*TTX);   // t2 f32 out (for t2s)
  float* FT   = alloc((size_t)BB*TFR*TTX);   // reused as viterbi Q
  float* LOGP = alloc((size_t)BB*TFR*TTX);
  float* LLS  = alloc((size_t)BB*TFR*257);
  float* XF   = alloc((size_t)BB*CCH*TFR);
  float* XP   = alloc((size_t)BB*CCH*TFR);
  _Float16* Ht  = allocH((size_t)BB*TFR*CCH);  // dwln out; reused as F1t
  _Float16* H1t = allocH((size_t)BB*TFR*CCH);  // pw1 out; reused as F2t
  _Float16* xth = allocH((size_t)BB*TTX*CCH);
  _Float16* yth = allocH((size_t)BB*TFR*ODIMP);
  _Float16* THt = allocH((size_t)BB*TTX*CCH);
  _Float16* Ttc = allocH((size_t)BB*TTX*CCH);
  _Float16* Ftt = allocH((size_t)BB*TFR*CCH);
  _Float16* WH  = allocH((size_t)8*1024*1024);
  float* f2s  = alloc((size_t)BB*TFR);
  float* t2s  = alloc((size_t)BB*TTX);
  float* dur  = alloc((size_t)BB*TTX);
  float* cum  = alloc((size_t)BB*TTX);
  float* dpout= alloc((size_t)BB*TTX);
  float* pred = alloc((size_t)BB*2*TFR);
  float* binacc = alloc(BB);
  float* ctcl   = alloc(BB);
  int* Abuf = (int*)alloc((size_t)BB*TFR);
  int* idxb = (int*)alloc((size_t)BB*TFR);

  // ---- weight arena: f16 [tap][O][Ipad] ----
  size_t wo = 0;
  auto wslot = [&](const float* src, int O, int I, int kw, int Ipad) {
    _Float16* p = WH + wo;
    size_t n = (size_t)kw * O * Ipad;
    wo = (wo + n + 15) & ~(size_t)15;
    wconv_kernel<<<((int)n + 255) / 256, 256, 0, stream>>>(src, p, O, I, kw, Ipad);
    return p;
  };
  _Float16 *WHdp1[4], *WHdp2[4], *WHpp1[4], *WHpp2[4];
  for (int l = 0; l < 4; ++l) {
    WHdp1[l] = wslot(dpl[l].pw1w, CCH, CCH, 1, CCH);
    WHdp2[l] = wslot(dpl[l].pw2w, CCH, CCH, 1, CCH);
  }
  for (int l = 0; l < 4; ++l) {
    WHpp1[l] = wslot(ppl[l].pw1w, CCH, CCH, 1, CCH);
    WHpp2[l] = wslot(ppl[l].pw2w, CCH, CCH, 1, CCH);
  }
  _Float16* WHt1 = wslot(t1w, CCH, CCH, 3, CCH);
  _Float16* WHt2 = wslot(t2w, CCH, CCH, 1, CCH);
  _Float16* WHf1 = wslot(f1w, CCH, ODIMC, 3, ODIMP);
  _Float16* WHf2 = wslot(f2w, CCH, CCH, 3, CCH);
  _Float16* WHf3 = wslot(f3w, CCH, CCH, 1, CCH);

  const dim3 gS(TTX/16, CCH/16, BB);
  const dim3 gL(TFR/16, CCH/16, BB);
  const int nXsm = BB*CCH*TTX, nXlg = BB*CCH*TFR;

  // transposed f16 copies of the raw inputs
  transpose_f16_kernel<<<dim3(TTX/16, CCH/16, BB), 256, 0, stream>>>(x, xth, CCH, TTX, CCH);
  transpose_f16_kernel<<<dim3(TFR/16, ODIMP/16, BB), 256, 0, stream>>>(y, yth, ODIMC, TFR, ODIMP);

  // ---- duration predictor (T=256) ----
  copy_kernel<<<(nXsm+255)/256, 256, 0, stream>>>(x, XDP, nXsm);
  for (int l = 0; l < 4; ++l) {
    const LayerP& P = dpl[l];
    dwln_kernel<<<BB*TTX, CCH, 0, stream>>>(XDP, P.dww, P.dwb, P.lng, P.lnb, Ht, TTX);
    conv_wmma2<CCH,1><<<gS, 32, 0, stream>>>(WHdp1[l], P.pw1b, Ht, nullptr, H1t,
                                             nullptr, nullptr, nullptr, CCH, TTX, 0, 1);
    conv_wmma2<CCH,1><<<gS, 32, 0, stream>>>(WHdp2[l], P.pw2b, H1t, XDP, nullptr,
                                             XDP, P.scale, xmask, CCH, TTX, 0, 0);
  }
  outconv_kernel<<<(BB*TTX+255)/256, 256, 0, stream>>>(dp_ow, dp_ob, XDP, dpout, xmask, 1, TTX);

  // ---- alignment module ----
  conv_wmma2<CCH,3><<<gS, 32, 0, stream>>>(WHt1, t1b, xth, nullptr, THt,
                                           nullptr, nullptr, nullptr, CCH, TTX, 1, 2);
  conv_wmma2<CCH,1><<<gS, 32, 0, stream>>>(WHt2, t2b, THt, Tb, Ttc,
                                           nullptr, nullptr, nullptr, CCH, TTX, 0, 0);
  conv_wmma2<ODIMP,3><<<gL, 32, 0, stream>>>(WHf1, f1b, yth, nullptr, Ht,
                                             nullptr, nullptr, nullptr, CCH, TFR, 1, 2);
  conv_wmma2<CCH,3><<<gL, 32, 0, stream>>>(WHf2, f2b, Ht, nullptr, H1t,
                                           nullptr, nullptr, nullptr, CCH, TFR, 1, 2);
  conv_wmma2<CCH,1><<<gL, 32, 0, stream>>>(WHf3, f3b, H1t, Hb, Ftt,
                                           nullptr, nullptr, nullptr, CCH, TFR, 0, 0);
  sumsq_kernel<<<(BB*TFR+255)/256, 256, 0, stream>>>(Hb, f2s, TFR);
  sumsq_kernel<<<(BB*TTX+255)/256, 256, 0, stream>>>(Tb, t2s, TTX);
  ft_wmma2<<<dim3(TTX/16, TFR/16, BB), 32, 0, stream>>>(Ftt, Ttc, FT);
  score_kernel<<<BB*TFR, TTX, 0, stream>>>(FT, f2s, t2s, xlen, LOGP);

  // ---- viterbi (Q reuses FT) ----
  vit_fwd_kernel<<<BB, TTX, 0, stream>>>(LOGP, FT);
  vit_back_kernel<<<1, BB, 0, stream>>>(FT, xlen, ylen, Abuf);
  dur_kernel<<<(BB*TTX+255)/256, 256, 0, stream>>>(Abuf, ylen, dur);
  binacc_kernel<<<1, BB, 0, stream>>>(Abuf, LOGP, ylen, binacc);

  // ---- CTC forward loss ----
  lls_kernel<<<BB*TFR, TTX, 0, stream>>>(LOGP, xlen, LLS);
  ctc_kernel<<<BB, 544, 0, stream>>>(LLS, xlen, ylen, ctcl);

  // ---- path expansion + gather ----
  cum_kernel<<<1, BB, 0, stream>>>(dur, cum);
  assign_kernel<<<(BB*TFR+255)/256, 256, 0, stream>>>(cum, xmask, ymask, idxb);
  gather_kernel<<<(nXlg+255)/256, 256, 0, stream>>>(x, idxb, XF, XP);

  // ---- pitch predictor (T=1024) ----
  for (int l = 0; l < 4; ++l) {
    const LayerP& P = ppl[l];
    dwln_kernel<<<BB*TFR, CCH, 0, stream>>>(XP, P.dww, P.dwb, P.lng, P.lnb, Ht, TFR);
    conv_wmma2<CCH,1><<<gL, 32, 0, stream>>>(WHpp1[l], P.pw1b, Ht, nullptr, H1t,
                                             nullptr, nullptr, nullptr, CCH, TFR, 0, 1);
    conv_wmma2<CCH,1><<<gL, 32, 0, stream>>>(WHpp2[l], P.pw2b, H1t, XP, nullptr,
                                             XP, P.scale, ymask, CCH, TFR, 0, 0);
  }
  outconv_kernel<<<(BB*2*TFR+255)/256, 256, 0, stream>>>(pp_ow, pp_ob, XP, pred, ymask, 2, TFR);

  // ---- outputs ----
  float* out = (float*)d_out;
  out_xframe_kernel<<<(nXlg+255)/256, 256, 0, stream>>>(XF, pew, peb, lcf0, ymask, out);
  const int nsm = 2*BB*TTX + 2*BB*TFR;
  out_small_kernel<<<(nsm+255)/256, 256, 0, stream>>>(dur, dpout, pred, out);
  loss_kernel<<<1, BB, 0, stream>>>(binacc, ylen, ctcl, out);
}
